// WindLSTM_36575941493504
// MI455X (gfx1250) — matrix operand order repaired and checked by reading, not compiled, not run on hardware
//
#include <hip/hip_runtime.h>
#include <hip/hip_bf16.h>

// ---------------------------------------------------------------------------
// WindLSTM on MI455X (gfx1250): persistent WMMA-bf16 LSTM scan.
//   B=4, T=1024, F=64, H=(2048,1024,512,256), 4096 sequential steps.
//   Weights pre-cast to bf16 + pre-swizzled into WMMA A-fragment order.
//   Mat-vec: v_wmma_f32_16x16x32_bf16 with vector replicated across B cols.
// ---------------------------------------------------------------------------

typedef __attribute__((ext_vector_type(8)))  float          v8f;
typedef __attribute__((ext_vector_type(16))) __bf16         v16bf;
typedef __attribute__((ext_vector_type(16))) unsigned short v16u;

#define NWG   64      // persistent workgroups (device barrier participants)
#define NTH   256     // 8 wave32 per WG
#define NSTEP 4096
#define EPS   1e-5f

static __device__ const int cIN[4] = {  64, 2048, 1024, 512 };
static __device__ const int cH [4] = {2048, 1024,  512, 256 };
// Kt = IN+H = {2112, 3072, 1536, 768}; all multiples of 32.

__device__ __forceinline__ unsigned short f2bf(float f) {
    unsigned u = __builtin_bit_cast(unsigned, f);
    unsigned r = 0x7FFFu + ((u >> 16) & 1u);          // round-to-nearest-even
    return (unsigned short)((u + r) >> 16);
}
__device__ __forceinline__ float sigm(float x) {
    return 1.0f / (1.0f + __expf(-x));
}

__device__ __forceinline__ v8f wmma_bf16(v16u a_, v16bf b, v8f c) {
    v16bf a = __builtin_bit_cast(v16bf, a_);
    // 8 args: (neg_a, A, neg_b, B, c_mod, C, reuse_a, reuse_b)
    return __builtin_amdgcn_wmma_f32_16x16x32_bf16(false, a, false, b,
                                                   (short)0, c, false, false);
}

// ---------------- BatchNorm over (N,L) per timestep-channel ----------------
__global__ __launch_bounds__(256) void bn_kernel(const float* __restrict__ x,
                                                 const float* __restrict__ w,
                                                 const float* __restrict__ bb,
                                                 unsigned short* __restrict__ xn) {
    __shared__ float s1[256], s2[256];
    const int t = blockIdx.x, tid = threadIdx.x;
    const int b = tid >> 6, f = tid & 63;
    const size_t idx = (size_t)b * 65536u + (size_t)t * 64u + f;  // (b*T+t)*64+f
    float v = x[idx];
    s1[tid] = v; s2[tid] = v * v;
    __syncthreads();
    for (int s = 128; s > 0; s >>= 1) {
        if (tid < s) { s1[tid] += s1[tid + s]; s2[tid] += s2[tid + s]; }
        __syncthreads();
    }
    float mean = s1[0] * (1.0f / 256.0f);
    float var  = s2[0] * (1.0f / 256.0f) - mean * mean;   // biased var
    float rs   = rsqrtf(var + EPS);
    float xv   = (v - mean) * rs * w[t] + bb[t];
    xn[idx] = f2bf(xv);
}

// -------- pre-swizzle [W_ih | W_hh] (fp32 row-major) -> WMMA-A bf16 --------
// A 16x32 bf16 fragment, k order used here (B matches): lane m (0-15) holds row m, K={0..15};
// lane m+16 holds row m, K={16..31}; element s of the lane is K=16*half+s.
__global__ __launch_bounds__(256) void swz_kernel(const float* __restrict__ wih,
                                                  const float* __restrict__ whh,
                                                  unsigned short* __restrict__ dst,
                                                  int IN, int H, long long total) {
    long long e = (long long)blockIdx.x * 256 + threadIdx.x;
    const long long stride = (long long)gridDim.x * 256;
    const int nk = (IN + H) >> 5;
    for (; e < total; e += stride) {
        int       r    = (int)(e & 511);
        long long blk  = e >> 9;
        int lane = r >> 4, idxv = r & 15;
        int kt = (int)(blk % nk), mt = (int)(blk / nk);
        int m    = mt * 16 + (lane & 15);
        int half = lane >> 4;
        int r2 = idxv >> 1, b0 = idxv & 1;
        int k = kt * 32 + 16 * half + idxv;
        float v = (k < IN) ? wih[(long long)m * IN + k]
                           : whh[(long long)m * H + (k - IN)];
        dst[e] = f2bf(v);
    }
}

__global__ __launch_bounds__(256) void bias_kernel(const float* __restrict__ bi,
                                                   const float* __restrict__ bh,
                                                   float* __restrict__ dst, int n) {
    int i = blockIdx.x * 256 + threadIdx.x;
    if (i < n) dst[i] = bi[i] + bh[i];
}

// ------------------------------ persistent scan ----------------------------
struct LP {
    const v16u*     wsw[4];   // swizzled combined weights per layer
    const float*    bias[4];  // combined biases (4H)
    const unsigned short* xn; // normalized input, bf16, 4096*64
    unsigned short* iv[4];    // input vectors [2 parities][Kt] bf16, layout [x|h]
    float*          c[4];     // cell states
    float*          out;      // 4096*256 fp32
    unsigned*       cnt;      // barrier counter
    unsigned*       gen;      // barrier generation
};

__device__ __forceinline__ void gbar(unsigned* cnt, unsigned* gen) {
    __syncthreads();
    if (threadIdx.x == 0) {
        __threadfence();
        unsigned g = __hip_atomic_load(gen, __ATOMIC_ACQUIRE, __HIP_MEMORY_SCOPE_AGENT);
        unsigned a = __hip_atomic_fetch_add(cnt, 1u, __ATOMIC_ACQ_REL, __HIP_MEMORY_SCOPE_AGENT);
        if (a == NWG - 1) {
            __hip_atomic_store(cnt, 0u, __ATOMIC_RELAXED, __HIP_MEMORY_SCOPE_AGENT);
            __hip_atomic_fetch_add(gen, 1u, __ATOMIC_RELEASE, __HIP_MEMORY_SCOPE_AGENT);
        } else {
            while (__hip_atomic_load(gen, __ATOMIC_ACQUIRE, __HIP_MEMORY_SCOPE_AGENT) == g)
                __builtin_amdgcn_s_sleep(2);
        }
    }
    __syncthreads();
}

__global__ __launch_bounds__(NTH) void lstm_scan(LP p) {
    __shared__ unsigned s_vec[1536];          // up to Kt=3072 bf16 = 6 KB
    __shared__ float    s_part[8][4][16];     // per-wave partial gate sums
    __shared__ float    s_sums[4][16];

    const int tid  = threadIdx.x;
    const int w    = tid >> 5;
    const int lane = tid & 31;
    const int wg   = blockIdx.x;

    // ---- init: zero c and h-parity0, seed x-part of layer1 parity0 ----
    if (wg == 0) {
        for (int l = 0; l < 4; ++l) {
            const int IN = cIN[l], H = cH[l];
            for (int i = tid; i < H; i += NTH) {
                p.c[l][i] = 0.0f;
                p.iv[l][IN + i] = 0;          // parity-0 h part
            }
        }
        for (int i = tid; i < 64; i += NTH) p.iv[0][i] = p.xn[i];
    }

    for (int t = 0; t < NSTEP; ++t) {
        for (int l = 0; l < 4; ++l) {
            gbar(p.cnt, p.gen);               // inputs for (t,l) are now ready
            const int IN = cIN[l], H = cH[l];
            const int Kt = IN + H, nk = Kt >> 5, njt = H >> 4, H16 = H >> 4;

            // stage input vector [x_t | h_{t-1}] (bf16) into LDS
            {
                const unsigned* g32 =
                    (const unsigned*)(p.iv[l] + (size_t)(t & 1) * Kt);
                for (int i = tid; i < (Kt >> 1); i += NTH) s_vec[i] = g32[i];
            }
            __syncthreads();
            const unsigned short* sv = (const unsigned short*)s_vec;

            const int c0 = (nk * w) >> 3, c1 = (nk * (w + 1)) >> 3;

            for (int jt = wg; jt < njt; jt += NWG) {
                v8f a0 = {0,0,0,0,0,0,0,0}, a1 = a0, a2 = a0, a3 = a0;
                const v16u* p0 = p.wsw[l] + ((size_t)(0 * H16 + jt) * nk + c0) * 32 + lane;
                const v16u* p1 = p.wsw[l] + ((size_t)(1 * H16 + jt) * nk + c0) * 32 + lane;
                const v16u* p2 = p.wsw[l] + ((size_t)(2 * H16 + jt) * nk + c0) * 32 + lane;
                const v16u* p3 = p.wsw[l] + ((size_t)(3 * H16 + jt) * nk + c0) * 32 + lane;

                for (int kt = c0; kt < c1; ++kt) {
                    // B 32x16 bf16: lanes 0-15 K=k0..k0+15, lanes 16-31 K=+16
                    v16u  bt = *(const v16u*)(sv + (kt << 5) + ((lane >> 4) << 4));
                    v16bf bf = __builtin_bit_cast(v16bf, bt);
                    a0 = wmma_bf16(p0[0], bf, a0); p0 += 32;
                    a1 = wmma_bf16(p1[0], bf, a1); p1 += 32;
                    a2 = wmma_bf16(p2[0], bf, a2); p2 += 32;
                    a3 = wmma_bf16(p3[0], bf, a3); p3 += 32;
                }

                // D layout: lane0 col0 rows0-7 in v0..7, lane16 rows8-15
                __syncthreads();
                if (lane == 0 || lane == 16) {
                    const int rb = (lane >> 4) << 3;
#pragma unroll
                    for (int r = 0; r < 8; ++r) {
                        s_part[w][0][rb + r] = a0[r];
                        s_part[w][1][rb + r] = a1[r];
                        s_part[w][2][rb + r] = a2[r];
                        s_part[w][3][rb + r] = a3[r];
                    }
                }
                __syncthreads();
                if (tid < 64) {
                    const int g = tid >> 4, rr = tid & 15;
                    float s = p.bias[l][g * H + jt * 16 + rr];
#pragma unroll
                    for (int w2 = 0; w2 < 8; ++w2) s += s_part[w2][g][rr];
                    s_sums[g][rr] = s;
                }
                __syncthreads();
                if (tid < 16) {
                    const int j = jt * 16 + tid;
                    const float ig = sigm(s_sums[0][tid]);
                    const float fg = sigm(s_sums[1][tid]);
                    const float gg = tanhf(s_sums[2][tid]);
                    const float og = sigm(s_sums[3][tid]);
                    const float cn = fg * p.c[l][j] + ig * gg;
                    p.c[l][j] = cn;
                    const float h = og * tanhf(cn);
                    const unsigned short hb = f2bf(h);
                    // recurrent input for layer l at step t+1 (other parity)
                    p.iv[l][(size_t)((t + 1) & 1) * Kt + IN + j] = hb;
                    if (l < 3) {
                        // feedforward input of layer l+1 at step t (same parity)
                        const int Kt1 = cIN[l + 1] + cH[l + 1];
                        p.iv[l + 1][(size_t)(t & 1) * Kt1 + j] = hb;
                    } else {
                        p.out[(size_t)t * 256 + j] = h;
                    }
                }
                __syncthreads();
            }

            // during layer-4 phase, an idle WG stages xn row t+1 for layer 1
            if (l == 3 && wg == NWG - 1 && t + 1 < NSTEP) {
                const int Kt0 = cIN[0] + cH[0];
                for (int i = tid; i < 64; i += NTH)
                    p.iv[0][(size_t)((t + 1) & 1) * Kt0 + i] =
                        p.xn[(size_t)(t + 1) * 64 + i];
            }
        }
    }
}

// ------------------------------- host side ---------------------------------
extern "C" void kernel_launch(void* const* d_in, const int* in_sizes, int n_in,
                              void* d_out, int out_size, void* d_ws, size_t ws_size,
                              hipStream_t stream) {
    (void)in_sizes; (void)n_in; (void)out_size;

    static const long long WSWE[4] = {8192LL * 2112, 4096LL * 3072,
                                      2048LL * 1536, 1024LL * 768};
    static const int INs[4] = {64, 2048, 1024, 512};
    static const int Hs [4] = {2048, 1024, 512, 256};

    auto al = [](size_t x) { return (x + 255) & ~size_t(255); };
    char* ws = (char*)d_ws;
    size_t off = 0;
    size_t wswOff[4], biasOff[4], ivOff[4], cOff[4];
    for (int l = 0; l < 4; ++l) { wswOff[l] = off; off = al(off + (size_t)WSWE[l] * 2); }
    for (int l = 0; l < 4; ++l) { biasOff[l] = off; off = al(off + (size_t)4 * Hs[l] * 4); }
    size_t xnOff = off; off = al(off + (size_t)NSTEP * 64 * 2);
    for (int l = 0; l < 4; ++l) {
        ivOff[l] = off; off = al(off + (size_t)2 * (INs[l] + Hs[l]) * 2);
    }
    for (int l = 0; l < 4; ++l) { cOff[l] = off; off = al(off + (size_t)Hs[l] * 4); }
    size_t barOff = off; off = al(off + 16);
    if (off > ws_size) return;  // workspace too small: bail (output stays invalid)

    const float* x    = (const float*)d_in[0];
    const float* bn_w = (const float*)d_in[1];
    const float* bn_b = (const float*)d_in[2];

    // 1) BatchNorm -> bf16 normalized input
    bn_kernel<<<1024, 256, 0, stream>>>(x, bn_w, bn_b,
                                        (unsigned short*)(ws + xnOff));

    // 2) weight swizzle + bias combine per layer
    for (int l = 0; l < 4; ++l) {
        const float* wih = (const float*)d_in[3 + 4 * l];
        const float* whh = (const float*)d_in[4 + 4 * l];
        const float* bih = (const float*)d_in[5 + 4 * l];
        const float* bhh = (const float*)d_in[6 + 4 * l];
        swz_kernel<<<4096, 256, 0, stream>>>(wih, whh,
                                             (unsigned short*)(ws + wswOff[l]),
                                             INs[l], Hs[l], WSWE[l]);
        int n4h = 4 * Hs[l];
        bias_kernel<<<(n4h + 255) / 256, 256, 0, stream>>>(bih, bhh,
                                                           (float*)(ws + biasOff[l]), n4h);
    }

    // 3) barrier state
    hipMemsetAsync(ws + barOff, 0, 16, stream);

    // 4) persistent scan
    LP p;
    for (int l = 0; l < 4; ++l) {
        p.wsw[l]  = (const v16u*)(ws + wswOff[l]);
        p.bias[l] = (const float*)(ws + biasOff[l]);
        p.iv[l]   = (unsigned short*)(ws + ivOff[l]);
        p.c[l]    = (float*)(ws + cOff[l]);
    }
    p.xn  = (const unsigned short*)(ws + xnOff);
    p.out = (float*)d_out;
    p.cnt = (unsigned*)(ws + barOff);
    p.gen = (unsigned*)(ws + barOff + 8);

    lstm_scan<<<NWG, NTH, 0, stream>>>(p);
}
